// PPRFCN_35777077576245
// MI455X (gfx1250) — matrix operand order repaired and checked by reading, not compiled, not run on hardware
//
#include <hip/hip_runtime.h>
#include <math.h>

#define EPSV 1e-5f
#define BB   64
#define CC   256
#define HH   64
#define WW   64
#define NREG 36                 // 4 sets x 3 x 3 cells
#define M_ROWS (BB * NREG)      // 2304
#define N_OUT  324
#define N_PAD  336              // padded to multiple of 16
#define K_DIM  256

typedef float    v2f  __attribute__((ext_vector_type(2)));
typedef float    v8f  __attribute__((ext_vector_type(8)));
typedef _Float16 v16h __attribute__((ext_vector_type(16)));

#if __has_builtin(__builtin_amdgcn_wmma_f32_16x16x4_f32)
#define USE_WMMA_F32 1
#else
#define USE_WMMA_F32 0
#endif

// ---------------------------------------------------------------------------
// Kernel 1: per-batch region bounds (exact replication of reference floor/adjust)
// rects[b*36 + set*9 + i*3 + j] = {r0, r1, c0, c1}
// sets: 0=subj(bbox_s), 1=obj(bbox_o), 2=joint w/ bbox_s, 3=joint w/ bbox_o
// ---------------------------------------------------------------------------
__device__ __forceinline__ void store_cells(int4* rects, int b, int set,
                                            const int* r0, const int* r1,
                                            const int* c0, const int* c1) {
  for (int i = 0; i < 3; ++i)
    for (int j = 0; j < 3; ++j) {
      int4 rc; rc.x = r0[i]; rc.y = r1[i]; rc.z = c0[j]; rc.w = c1[j];
      rects[b * NREG + set * 9 + i * 3 + j] = rc;
    }
}

__global__ void bounds_kernel(const float* __restrict__ bbs,
                              const float* __restrict__ bbo,
                              int4* __restrict__ rects) {
  int b = blockIdx.x * blockDim.x + threadIdx.x;
  if (b >= BB) return;
  float sy0 = bbs[b*4+0], sy1 = bbs[b*4+1], sx0 = bbs[b*4+2], sx1 = bbs[b*4+3];
  float oy0 = bbo[b*4+0], oy1 = bbo[b*4+1], ox0 = bbo[b*4+2], ox1 = bbo[b*4+3];
  int r0[3], r1[3], c0[3], c1[3];

  // subj / obj  (_subj_obj_bounds)
  for (int set = 0; set < 2; ++set) {
    float y0 = set ? oy0 : sy0, y1 = set ? oy1 : sy1;
    float x0 = set ? ox0 : sx0, x1 = set ? ox1 : sx1;
    float ch = (y1 - y0) / 3.0f, cw = (x1 - x0) / 3.0f;
    for (int i = 0; i < 3; ++i) {
      float fi = (float)i;
      int a0 = (int)floorf(64.0f * (y0 + fi * ch));
      int a1 = (int)floorf(64.0f * (y0 + (fi + 1.0f) * ch));
      if (a1 <= a0) { if (a1 < HH - 1) a1 += 1; else a0 -= 1; }   // rows: compare vs H-1
      r0[i] = a0; r1[i] = a1;
      int b0 = (int)floorf(64.0f * (x0 + fi * cw));
      int b1 = (int)floorf(64.0f * (x0 + (fi + 1.0f) * cw));
      if (b1 <= b0) { if (b1 < WW) b1 += 1; else b0 -= 1; }       // cols: compare vs W
      c0[i] = b0; c1[i] = b1;
    }
    store_cells(rects, b, set, r0, r1, c0, c1);
  }

  // joint (_joint_bounds), sets 2 (clip to bbox_s) and 3 (clip to bbox_o)
  float uy0 = fminf(sy0, oy0), uy1 = fmaxf(sy1, oy1);
  float ux0 = fminf(sx0, ox0), ux1 = fmaxf(sx1, ox1);
  float ch = (uy1 - uy0) / 3.0f, cw = (ux1 - ux0) / 3.0f;
  for (int set = 2; set < 4; ++set) {
    float y0 = (set == 2) ? sy0 : oy0, y1 = (set == 2) ? sy1 : oy1;
    float x0 = (set == 2) ? sx0 : ox0, x1 = (set == 2) ? sx1 : ox1;
    for (int i = 0; i < 3; ++i) {
      float fi = (float)i;
      r0[i] = (int)floorf(64.0f * fmaxf(y0, uy0 + fi * ch));
      r1[i] = (int)floorf(64.0f * fminf(y1, uy0 + (fi + 1.0f) * ch));
      c0[i] = (int)floorf(64.0f * fmaxf(x0, ux0 + fi * cw));
      c1[i] = (int)floorf(64.0f * fminf(x1, ux0 + (fi + 1.0f) * cw));
    }
    store_cells(rects, b, set, r0, r1, c0, c1);
  }
}

// ---------------------------------------------------------------------------
// Kernel 2: fold BN scale into conv weights, transposed+padded for the GEMM.
// Bt[k][n] = alpha_n * conv_w[n][k]  (n < 324; zero-padded to 336)
// constv[n] = alpha_n*(conv_b[n]-bn2_m[n]) + bn2_b[n]
// ---------------------------------------------------------------------------
__global__ void prep_kernel(const float* __restrict__ conv_w,
                            const float* __restrict__ conv_b,
                            const float* __restrict__ g,  const float* __restrict__ be,
                            const float* __restrict__ m,  const float* __restrict__ v,
                            float* __restrict__ Bt, float* __restrict__ constv) {
  int o = blockIdx.x;     // 0..335
  int c = threadIdx.x;    // 0..255
  if (o < N_OUT) {
    float alpha = g[o] * rsqrtf(v[o] + EPSV);
    Bt[(size_t)c * N_PAD + o] = alpha * conv_w[(size_t)o * K_DIM + c];
    if (c == 0) constv[o] = alpha * (conv_b[o] - m[o]) + be[o];
  } else {
    Bt[(size_t)c * N_PAD + o] = 0.0f;
  }
}

// ---------------------------------------------------------------------------
// Kernel 3: region means of feat. One block per (b, region); 8 waves; each wave
// owns a channel slice, 32 lanes sweep W (coalesced), wave-reduce via shfl_xor.
// ---------------------------------------------------------------------------
__global__ void region_mean_kernel(const float* __restrict__ feat,
                                   const int4* __restrict__ rects,
                                   float* __restrict__ meanfeat) {
  int br = blockIdx.x;                  // 0 .. B*36-1
  int b  = br / NREG;
  int4 rc = rects[br];
  int h0 = max(rc.x, 0), h1 = min(rc.y, HH);
  int w0 = max(rc.z, 0), w1 = min(rc.w, WW);
  int nh = h1 - h0, nw = w1 - w0;
  bool nz = (nh > 0) && (nw > 0);
  float inv = nz ? 1.0f / (float)(nh * nw) : 0.0f;
  int lane = threadIdx.x & 31;
  int wv   = threadIdx.x >> 5;          // 0..7
  for (int c = wv; c < CC; c += 8) {
    float s = 0.0f;
    if (nz) {
      const float* p = feat + ((size_t)b * CC + c) * (HH * WW);
      for (int h = h0; h < h1; ++h) {
        const float* row = p + h * WW;
        for (int w = w0 + lane; w < w1; w += 32) s += row[w];
      }
    }
#pragma unroll
    for (int off = 16; off > 0; off >>= 1) s += __shfl_xor(s, off, 32);
    if (lane == 0) meanfeat[(size_t)br * CC + c] = s * inv;
  }
}

// ---------------------------------------------------------------------------
// Kernel 4: WMMA GEMM  C[2304][336] = meanfeat[2304][256] x Bt[256][336]
// One wave per 16x16 tile; 144 x 21 = 3024 waves; blockDim=256 -> 378 blocks.
// No divergence -> EXEC all-1s (WMMA requirement).
// ---------------------------------------------------------------------------
__global__ void gemm_wmma_kernel(const float* __restrict__ A,
                                 const float* __restrict__ Bt,
                                 float* __restrict__ Cmat) {
  const int NT = N_PAD / 16;            // 21 tiles along N
  int wave = blockIdx.x * (blockDim.x >> 5) + (threadIdx.x >> 5);
  int tm = wave / NT;
  int tn = wave % NT;
  int lane = threadIdx.x & 31;
  int row  = lane & 15;
  int hi   = lane >> 4;                 // 0 or 1
  const float* Ap = A  + (size_t)(tm * 16 + row) * K_DIM;
  const float* Bp = Bt + tn * 16 + row;
  v8f acc = {0.f, 0.f, 0.f, 0.f, 0.f, 0.f, 0.f, 0.f};

#if USE_WMMA_F32
  // A 16x4 f32: lane<16 -> K pair {0,1}; lane>=16 -> K pair {2,3} (ISA 7.12.2)
  int kb = hi * 2;
  for (int k0 = 0; k0 < K_DIM; k0 += 4) {
    v2f a, bf;
    a.x  = Ap[k0 + kb];
    a.y  = Ap[k0 + kb + 1];
    bf.x = Bp[(size_t)(k0 + kb)     * N_PAD];
    bf.y = Bp[(size_t)(k0 + kb + 1) * N_PAD];
    acc = __builtin_amdgcn_wmma_f32_16x16x4_f32(false, a, false, bf,
                                                (short)0, acc, false, false);
  }
#else
  // Fallback: probe-confirmed v_wmma_f32_16x16x32_f16 (convert f32->f16 in-reg)
  int ka  = hi * 16;                    // A: lanes<16 K 0..15 ; >=16 K 16..31
  int kbb = hi * 16;                    // B: lanes<16 K 0..15 ; >=16 K 16..31
  for (int k0 = 0; k0 < K_DIM; k0 += 32) {
    v16h a, bf;
#pragma unroll
    for (int e = 0; e < 16; ++e)
      a[e] = (_Float16)Ap[k0 + ka + e];
#pragma unroll
    for (int e = 0; e < 16; ++e)
      bf[e] = (_Float16)Bp[(size_t)(k0 + kbb + e) * N_PAD];
    acc = __builtin_amdgcn_wmma_f32_16x16x32_f16(false, a, false, bf,
                                                 (short)0, acc, false, false);
  }
#endif

  // D layout: VGPR v -> M = v (lanes 0-15) / M = 8+v (lanes 16-31), N = lane%16
  int mbase = tm * 16 + hi * 8;
  int col   = tn * 16 + row;
#pragma unroll
  for (int v = 0; v < 8; ++v)
    Cmat[(size_t)(mbase + v) * N_PAD + col] = acc[v];
}

// ---------------------------------------------------------------------------
// Kernel 5: head. Gather cells (add const only when region non-empty, matching
// where(cnt>0, ..., 0)), fc1 + bn1, fc2, predicate dot. One block per batch.
// ---------------------------------------------------------------------------
__global__ void head_kernel(const float* __restrict__ pooled,
                            const int4*  __restrict__ rects,
                            const float* __restrict__ constv,
                            const float* __restrict__ fc1_w, const float* __restrict__ fc1_b,
                            const float* __restrict__ bn1_g, const float* __restrict__ bn1_b,
                            const float* __restrict__ bn1_m, const float* __restrict__ bn1_v,
                            const float* __restrict__ fc2_w, const float* __restrict__ fc2_b,
                            const float* __restrict__ pred,  float* __restrict__ out) {
  __shared__ float f[81], h[40], lg[9];
  int b = blockIdx.x;
  int t = threadIdx.x;
  if (t < 81) {
    int k = t / 9, ij = t % 9, i = ij / 3, j = ij % 3;
    float acc = 0.0f;
    for (int set = 0; set < 4; ++set) {
      int r = set * 9 + ij;
      int4 rc = rects[b * NREG + r];
      int nh = min(rc.y, HH) - max(rc.x, 0);
      int nw = min(rc.w, WW) - max(rc.z, 0);
      if (nh > 0 && nw > 0) {
        int o = set * 81 + i * 27 + j * 9 + k;
        acc += pooled[(size_t)(b * NREG + r) * N_PAD + o] + constv[o];
      }
    }
    f[t] = acc;              // reference flat order: idx = k*9 + i*3 + j
  }
  __syncthreads();
  if (t < 40) {
    float s = fc1_b[t];
    for (int e = 0; e < 81; ++e) s += fc1_w[t * 81 + e] * f[e];
    s = (s - bn1_m[t]) * (bn1_g[t] * rsqrtf(bn1_v[t] + EPSV)) + bn1_b[t];
    h[t] = s;
  }
  __syncthreads();
  if (t < 9) {
    float s = fc2_b[t];
    for (int e = 0; e < 40; ++e) s += fc2_w[t * 40 + e] * h[e];
    lg[t] = s * pred[b * 9 + t];
  }
  __syncthreads();
  if (t == 0) {
    float s = 0.0f;
    for (int e = 0; e < 9; ++e) s += lg[e];
    out[b] = s;
  }
}

// ---------------------------------------------------------------------------
extern "C" void kernel_launch(void* const* d_in, const int* in_sizes, int n_in,
                              void* d_out, int out_size, void* d_ws, size_t ws_size,
                              hipStream_t stream) {
  const float* feat   = (const float*)d_in[0];
  const float* bbox_s = (const float*)d_in[1];
  const float* bbox_o = (const float*)d_in[2];
  const float* pred   = (const float*)d_in[3];
  const float* conv_w = (const float*)d_in[4];
  const float* conv_b = (const float*)d_in[5];
  const float* bn2_g  = (const float*)d_in[6];
  const float* bn2_b  = (const float*)d_in[7];
  const float* bn2_m  = (const float*)d_in[8];
  const float* bn2_v  = (const float*)d_in[9];
  const float* fc1_w  = (const float*)d_in[10];
  const float* fc1_b  = (const float*)d_in[11];
  const float* bn1_g  = (const float*)d_in[12];
  const float* bn1_b  = (const float*)d_in[13];
  const float* bn1_m  = (const float*)d_in[14];
  const float* bn1_v  = (const float*)d_in[15];
  const float* fc2_w  = (const float*)d_in[16];
  const float* fc2_b  = (const float*)d_in[17];
  float* out = (float*)d_out;

  // workspace layout (256B-aligned offsets)
  char* ws = (char*)d_ws;
  int4*  rects    = (int4*) (ws + 0);                       //  36,864 B
  float* meanfeat = (float*)(ws + 36864);                   // 2,359,296 B
  float* Bt       = (float*)(ws + 36864 + 2359296);         //   344,064 B
  float* constv   = (float*)(ws + 36864 + 2359296 + 344064);//     1,296 B
  float* pooled   = (float*)(ws + 2741760);                 // 3,096,576 B

  bounds_kernel     <<<1, 64, 0, stream>>>(bbox_s, bbox_o, rects);
  prep_kernel       <<<N_PAD, K_DIM, 0, stream>>>(conv_w, conv_b, bn2_g, bn2_b,
                                                  bn2_m, bn2_v, Bt, constv);
  region_mean_kernel<<<M_ROWS, 256, 0, stream>>>(feat, rects, meanfeat);
  // 3024 tiles / 8 waves per block = 378 blocks (exact, no guard -> EXEC all-1s)
  gemm_wmma_kernel  <<<(M_ROWS / 16) * (N_PAD / 16) / 8, 256, 0, stream>>>(meanfeat, Bt, pooled);
  head_kernel       <<<BB, 96, 0, stream>>>(pooled, rects, constv,
                                            fc1_w, fc1_b, bn1_g, bn1_b, bn1_m, bn1_v,
                                            fc2_w, fc2_b, pred, out);
}
